// SimpleGNN_86277303042435
// MI455X (gfx1250) — matrix operand order repaired and checked by reading, not compiled, not run on hardware
//
#include <hip/hip_runtime.h>
#include <hip/hip_bf16.h>

typedef __attribute__((ext_vector_type(16))) __bf16 v16bf;
typedef __attribute__((ext_vector_type(8)))  float  v8f;

#define KF 1081   // F_IN
#define NGR 64    // graphs

__device__ __forceinline__ void atomAddF(float* p, float v) {
  __hip_atomic_fetch_add(p, v, __ATOMIC_RELAXED, __HIP_MEMORY_SCOPE_AGENT);
}

// ---------------- degree / norm ----------------
__global__ void k_deg_init(float* __restrict__ degf, int n) {
  int i = blockIdx.x * 256 + threadIdx.x;
  if (i < n) degf[i] = 1.0f;              // self loop
}

__global__ void k_deg_edge(const int* __restrict__ dst, float* __restrict__ degf, int E) {
  int e = blockIdx.x * 256 + threadIdx.x;
  if (e < E) atomAddF(&degf[dst[e]], 1.0f);
}

__global__ void k_dinv(const float* __restrict__ degf, float* __restrict__ dinv, int n) {
  int i = blockIdx.x * 256 + threadIdx.x;
  if (i < n) dinv[i] = rsqrtf(degf[i]);   // deg >= 1 always
}

// ---------------- GEMM1: hw1[n,32] = x[n,1081] @ W1[1081,32]  (bf16 WMMA) ----------------
// 256 threads = 8 waves; each wave owns a 16-row tile, full 32 output cols (2 WMMA accums).
// K padded to 1088 = 34 chunks of 32; W1 staged in LDS as B-fragments in 2 phases of 17.
// Chunks 0..32 are fully in-bounds (33*32 = 1056 <= 1081) -> branch-free steady state;
// chunk 33 is peeled and handled with static masking after the phase-1 loop.
__global__ __launch_bounds__(256) void k_gemm1(const float* __restrict__ x,
                                               const float* __restrict__ W1,
                                               float* __restrict__ hw1, int n) {
  __shared__ __align__(32) __bf16 ldsB[17 * 2 * 32 * 16];   // 34816 B
  const int tid  = threadIdx.x;
  const int wave = tid >> 5;
  const int lane = tid & 31;
  const int m    = lane & 15;      // A row within tile / D col
  const int h    = lane >> 4;      // lane half
  const int r0   = (blockIdx.x * 8 + wave) * 16;
  int row = r0 + m; if (row >= n) row = n - 1;   // clamp (stores guarded)
  const float* __restrict__ px = x + (long)row * KF;

  v8f acc0 = {}; v8f acc1 = {};

  for (int phase = 0; phase < 2; ++phase) {
    __syncthreads();
    // Fill B fragments for chunks [phase*17, phase*17+17)
    for (int idx = tid; idx < 17 * 2 * 32 * 16; idx += 256) {
      int e = idx & 15;
      int l = (idx >> 4) & 31;
      int t = (idx >> 9) & 1;
      int c = (idx >> 10) + phase * 17;
      int col = t * 16 + (l & 15);
      int k   = c * 32 + (l >> 4) * 8 + (e & 7) + (e >> 3) * 16;
      ldsB[idx] = (__bf16)((k < KF) ? W1[k * 32 + col] : 0.0f);
    }
    __syncthreads();

    const int nfull = (phase == 0) ? 17 : 16;   // chunk 33 peeled below
    for (int cc = 0; cc < nfull; ++cc) {
      const int kb = (phase * 17 + cc) * 32;
      __builtin_prefetch(&px[kb + 32], 0, 1);   // next chunk of this lane's row
      v16bf a;
#pragma unroll
      for (int j = 0; j < 8; ++j) {
        a[j]     = (__bf16)px[kb + h * 8 + j];
        a[j + 8] = (__bf16)px[kb + 16 + h * 8 + j];
      }
      v16bf b0 = *(const v16bf*)&ldsB[((cc * 2 + 0) * 32 + lane) * 16];
      v16bf b1 = *(const v16bf*)&ldsB[((cc * 2 + 1) * 32 + lane) * 16];
      acc0 = __builtin_amdgcn_wmma_f32_16x16x32_bf16(false, a, false, b0, (short)0, acc0, false, false);
      acc1 = __builtin_amdgcn_wmma_f32_16x16x32_bf16(false, a, false, b1, (short)0, acc1, false, false);
    }
  }

  // Peeled tail: chunk 33 (kb = 1056); its B fragments are phase-1 slot cc=16.
  {
    const int kb = 33 * 32;
    v16bf a;
#pragma unroll
    for (int j = 0; j < 8; ++j) {
      int k1 = kb + h * 8 + j;        // 1056..1071 : always valid
      int k2 = k1 + 16;               // 1072..1087 : needs mask at >= 1081
      a[j]     = (__bf16)px[k1];
      a[j + 8] = (k2 < KF) ? (__bf16)px[k2] : (__bf16)0.0f;
    }
    v16bf b0 = *(const v16bf*)&ldsB[((16 * 2 + 0) * 32 + lane) * 16];
    v16bf b1 = *(const v16bf*)&ldsB[((16 * 2 + 1) * 32 + lane) * 16];
    acc0 = __builtin_amdgcn_wmma_f32_16x16x32_bf16(false, a, false, b0, (short)0, acc0, false, false);
    acc1 = __builtin_amdgcn_wmma_f32_16x16x32_bf16(false, a, false, b1, (short)0, acc1, false, false);
  }

  // D layout: lane -> (M = h*8 + r, N = m); lanes 0-15 per r give 64B-contiguous stores
#pragma unroll
  for (int r = 0; r < 8; ++r) {
    int rr = r0 + h * 8 + r;
    if (rr < n) {
      hw1[(long)rr * 32 + m]      = acc0[r];
      hw1[(long)rr * 32 + 16 + m] = acc1[r];
    }
  }
}

// ---------------- self-loop init: agg[i] = hw[i] * dinv[node]^2 ----------------
__global__ void k_self(const float* __restrict__ hw, const float* __restrict__ dinv,
                       float* __restrict__ agg, long total, int logF) {
  long i = (long)blockIdx.x * 256 + threadIdx.x;
  if (i < total) {
    int node = (int)(i >> logF);
    float dv = dinv[node];
    agg[i] = hw[i] * dv * dv;
  }
}

// ---------------- edge scatter, 32 features: one wave per edge ----------------
__global__ __launch_bounds__(256) void k_edge32(const int* __restrict__ src,
                                                const int* __restrict__ dst,
                                                const float* __restrict__ dinv,
                                                const float* __restrict__ hw,
                                                float* __restrict__ agg, int E) {
  int wave = threadIdx.x >> 5, lane = threadIdx.x & 31;
  int e = blockIdx.x * 8 + wave;
  if (e >= E) return;
  int s = src[e], d = dst[e];
  float nrm = dinv[s] * dinv[d];
  float v = hw[(long)s * 32 + lane] * nrm;
  atomAddF(&agg[(long)d * 32 + lane], v);
}

// ---------------- GEMM2: hw2[n,8] = relu(agg1 + b1) @ W2[32,8]  (one WMMA / 16 rows) ----------------
__global__ __launch_bounds__(256) void k_gemm2(const float* __restrict__ agg1,
                                               const float* __restrict__ b1,
                                               const float* __restrict__ W2,
                                               float* __restrict__ hw2, int n) {
  const int tid = threadIdx.x, wave = tid >> 5, lane = tid & 31;
  const int m = lane & 15, h = lane >> 4;
  const int r0 = (blockIdx.x * 8 + wave) * 16;
  int row = r0 + m; if (row >= n) row = n - 1;

  // B fragment: K = h*16+e (0..31), col = m; cols 8..15 zero-padded
  v16bf b;
#pragma unroll
  for (int e = 0; e < 16; ++e) {
    int k = h * 16 + e;
    b[e] = (__bf16)((m < 8) ? W2[k * 8 + m] : 0.0f);
  }
  const float* __restrict__ pa = agg1 + (long)row * 32;
  v16bf a;
#pragma unroll
  for (int j = 0; j < 8; ++j) {
    int f1 = h * 16 + j, f2 = h * 16 + 8 + j;
    float v1 = pa[f1] + b1[f1]; v1 = v1 > 0.0f ? v1 : 0.0f;
    float v2 = pa[f2] + b1[f2]; v2 = v2 > 0.0f ? v2 : 0.0f;
    a[j]     = (__bf16)v1;
    a[j + 8] = (__bf16)v2;
  }
  v8f acc = {};
  acc = __builtin_amdgcn_wmma_f32_16x16x32_bf16(false, a, false, b, (short)0, acc, false, false);
#pragma unroll
  for (int r = 0; r < 8; ++r) {
    int rr = r0 + h * 8 + r;
    if (rr < n && m < 8) hw2[(long)rr * 8 + m] = acc[r];
  }
}

// ---------------- edge scatter, 8 features: 4 edges per wave ----------------
__global__ __launch_bounds__(256) void k_edge8(const int* __restrict__ src,
                                               const int* __restrict__ dst,
                                               const float* __restrict__ dinv,
                                               const float* __restrict__ hw,
                                               float* __restrict__ agg, int E) {
  int wave = threadIdx.x >> 5, lane = threadIdx.x & 31;
  int sub = lane >> 3, f = lane & 7;
  long e = ((long)blockIdx.x * 8 + wave) * 4 + sub;
  if (e >= E) return;
  int s = src[e], d = dst[e];
  float nrm = dinv[s] * dinv[d];
  atomAddF(&agg[(long)d * 8 + f], hw[(long)s * 8 + f] * nrm);
}

// ---------------- pooling ----------------
__global__ void k_pool_zero(float* __restrict__ psum, float* __restrict__ pcnt) {
  int t = threadIdx.x;
  if (t < NGR * 8) psum[t] = 0.0f;
  if (t < NGR)     pcnt[t] = 0.0f;
}

__global__ __launch_bounds__(256) void k_pool(const float* __restrict__ agg2,
                                              const float* __restrict__ b2,
                                              const int* __restrict__ batch,
                                              float* __restrict__ psum,
                                              float* __restrict__ pcnt, int n) {
  __shared__ float lsum[NGR * 8];
  __shared__ float lcnt[NGR];
  int tid = threadIdx.x;
  for (int i = tid; i < NGR * 8; i += 256) lsum[i] = 0.0f;
  if (tid < NGR) lcnt[tid] = 0.0f;
  __syncthreads();
  int node = blockIdx.x * 256 + tid;
  if (node < n) {
    int g = batch[node];
    atomicAdd(&lcnt[g], 1.0f);
#pragma unroll
    for (int f = 0; f < 8; ++f) {
      float v = agg2[(long)node * 8 + f] + b2[f];
      v = v > 0.0f ? v : 0.0f;
      atomicAdd(&lsum[g * 8 + f], v);
    }
  }
  __syncthreads();
  for (int i = tid; i < NGR * 8; i += 256) atomAddF(&psum[i], lsum[i]);
  if (tid < NGR) atomAddF(&pcnt[tid], lcnt[tid]);
}

__global__ void k_final(const float* __restrict__ psum, const float* __restrict__ pcnt,
                        const float* __restrict__ Wfc, const float* __restrict__ bfc,
                        float* __restrict__ out) {
  int t = threadIdx.x;
  if (t < NGR * 5) {
    int g = t / 5, j = t % 5;
    float cnt = pcnt[g]; if (cnt < 1.0f) cnt = 1.0f;
    float s = 0.0f;
#pragma unroll
    for (int f = 0; f < 8; ++f) s += (psum[g * 8 + f] / cnt) * Wfc[f * 5 + j];
    out[t] = s + bfc[j];
  }
}

extern "C" void kernel_launch(void* const* d_in, const int* in_sizes, int n_in,
                              void* d_out, int out_size, void* d_ws, size_t ws_size,
                              hipStream_t stream) {
  const float* x   = (const float*)d_in[0];
  const float* W1  = (const float*)d_in[1];
  const float* b1  = (const float*)d_in[2];
  const float* W2  = (const float*)d_in[3];
  const float* b2  = (const float*)d_in[4];
  const float* Wfc = (const float*)d_in[5];
  const float* bfc = (const float*)d_in[6];
  const int*   ei  = (const int*)d_in[7];   // [2, E]
  const int*   bat = (const int*)d_in[8];   // [n]
  float* out = (float*)d_out;

  const int n = in_sizes[0] / KF;
  const int E = in_sizes[7] / 2;
  const int* srcp = ei;
  const int* dstp = ei + E;

  // workspace layout (floats)
  float* ws   = (float*)d_ws;
  float* degf = ws;                          // n
  float* dinv = ws + (size_t)n;              // n
  float* hw1  = ws + (size_t)2 * n;          // n*32
  float* agg1 = ws + (size_t)34 * n;         // n*32
  float* hw2  = ws + (size_t)66 * n;         // n*8
  float* agg2 = ws + (size_t)74 * n;         // n*8
  float* psum = ws + (size_t)82 * n;         // 512
  float* pcnt = psum + NGR * 8;              // 64

  const int g256n  = (n + 255) / 256;
  const int g256e  = (E + 255) / 256;
  const int gGemm  = (n + 127) / 128;        // 8 waves x 16 rows per block

  // 1) degree + norm
  k_deg_init<<<g256n, 256, 0, stream>>>(degf, n);
  k_deg_edge<<<g256e, 256, 0, stream>>>(dstp, degf, E);
  k_dinv<<<g256n, 256, 0, stream>>>(degf, dinv, n);

  // 2) layer 1: GEMM (WMMA bf16) -> self loop -> edge scatter
  k_gemm1<<<gGemm, 256, 0, stream>>>(x, W1, hw1, n);
  k_self<<<((long)n * 32 + 255) / 256, 256, 0, stream>>>(hw1, dinv, agg1, (long)n * 32, 5);
  k_edge32<<<(E + 7) / 8, 256, 0, stream>>>(srcp, dstp, dinv, hw1, agg1, E);

  // 3) layer 2: fused relu+bias into WMMA A-fragment -> self loop -> edge scatter
  k_gemm2<<<gGemm, 256, 0, stream>>>(agg1, b1, W2, hw2, n);
  k_self<<<((long)n * 8 + 255) / 256, 256, 0, stream>>>(hw2, dinv, agg2, (long)n * 8, 3);
  k_edge8<<<(E + 31) / 32, 256, 0, stream>>>(srcp, dstp, dinv, hw2, agg2, E);

  // 4) mean pool (LDS combine) + linear head
  k_pool_zero<<<1, 512, 0, stream>>>(psum, pcnt);
  k_pool<<<g256n, 256, 0, stream>>>(agg2, b2, bat, psum, pcnt, n);
  k_final<<<1, 320, 0, stream>>>(psum, pcnt, Wfc, bfc, out);
}
